// INLVisionBlock_40578851012932
// MI455X (gfx1250) — matrix operand order repaired and checked by reading, not compiled, not run on hardware
//
#include <hip/hip_runtime.h>

// ---------------------------------------------------------------------------
// Types for gfx1250 WMMA (wave32). bf16 inputs, f32 accumulate.
// ---------------------------------------------------------------------------
typedef __attribute__((ext_vector_type(16))) __bf16 v16bf;
typedef __attribute__((ext_vector_type(8)))  float  v8f;
typedef __attribute__((ext_vector_type(4)))  unsigned int v4u;

union Frag16 { v16bf v; v4u q[2]; };
union Chunk8 { v4u q; unsigned short s[8]; };

#define NTOK   12608   // B*S = 64*197
#define DMODEL 768
#define DQKV   2304
#define DFF    3072
#define SEQ    197
#define BATCH  64
#define NHEAD  12

__device__ __forceinline__ unsigned short f2bf(float f) {
  unsigned int u = __float_as_uint(f);
  u += 0x7fffu + ((u >> 16) & 1u);   // round-to-nearest-even
  return (unsigned short)(u >> 16);
}
__device__ __forceinline__ float gelu_exact(float v) {
  return 0.5f * v * (1.0f + erff(v * 0.70710678118654752f));
}
__device__ __forceinline__ v8f wmma_bf16(v16bf a, v16bf b, v8f c) {
  return __builtin_amdgcn_wmma_f32_16x16x32_bf16(false, a, false, b, (short)0, c,
                                                 false, false);
}
// CDNA5 async memory->LDS copy (16B per enabled lane), tracked by ASYNCcnt.
__device__ __forceinline__ void async_b128(const void* gptr, const void* lptr) {
  unsigned long long ga = (unsigned long long)(uintptr_t)gptr;
  unsigned int la = (unsigned int)(uintptr_t)lptr;  // low 32 bits = LDS byte addr
  asm volatile("global_load_async_to_lds_b128 %0, %1, off"
               :: "v"(la), "v"(ga) : "memory");
}
__device__ __forceinline__ void wait_async0() {
  asm volatile("s_wait_asynccnt 0x0" ::: "memory");
}

// ---------------------------------------------------------------------------
// float -> bf16 convert (weights, once per launch)
// ---------------------------------------------------------------------------
__global__ __launch_bounds__(256) void f2bf_kernel(const float* __restrict__ in,
                                                   unsigned short* __restrict__ out,
                                                   int n) {
  int i = blockIdx.x * 256 + threadIdx.x;
  if (i < n) out[i] = f2bf(in[i]);
}

__global__ __launch_bounds__(256) void add_kernel(const float* __restrict__ a,
                                                  const float* __restrict__ b,
                                                  float* __restrict__ c, int n) {
  int i = blockIdx.x * 256 + threadIdx.x;
  if (i < n) c[i] = a[i] + b[i];
}

// ---------------------------------------------------------------------------
// LayerNorm over D=768, one row per 256-thread block. bf16 out + optional f32.
// ---------------------------------------------------------------------------
__global__ __launch_bounds__(256) void ln_kernel(const float* __restrict__ x,
                                                 const float* __restrict__ g,
                                                 const float* __restrict__ bb,
                                                 unsigned short* __restrict__ outB,
                                                 float* __restrict__ outF) {
  __shared__ float red[256];
  const int tid = threadIdx.x;
  const size_t r = blockIdx.x;
  const float* xr = x + r * DMODEL;
  float v0 = xr[tid], v1 = xr[tid + 256], v2 = xr[tid + 512];
  red[tid] = v0 + v1 + v2;
  __syncthreads();
  for (int s = 128; s > 0; s >>= 1) { if (tid < s) red[tid] += red[tid + s]; __syncthreads(); }
  const float mu = red[0] * (1.0f / DMODEL);
  __syncthreads();
  float d0 = v0 - mu, d1 = v1 - mu, d2 = v2 - mu;
  red[tid] = d0 * d0 + d1 * d1 + d2 * d2;
  __syncthreads();
  for (int s = 128; s > 0; s >>= 1) { if (tid < s) red[tid] += red[tid + s]; __syncthreads(); }
  const float rstd = rsqrtf(red[0] * (1.0f / DMODEL) + 1e-5f);
  float y0 = d0 * rstd * g[tid]       + bb[tid];
  float y1 = d1 * rstd * g[tid + 256] + bb[tid + 256];
  float y2 = d2 * rstd * g[tid + 512] + bb[tid + 512];
  unsigned short* ob = outB + r * DMODEL;
  ob[tid] = f2bf(y0); ob[tid + 256] = f2bf(y1); ob[tid + 512] = f2bf(y2);
  if (outF) {
    float* of = outF + r * DMODEL;
    of[tid] = y0; of[tid + 256] = y1; of[tid + 512] = y2;
  }
}

// ---------------------------------------------------------------------------
// WMMA GEMM: C[M,E] = act(A[M,K] * W[E,K]^T + bias) * alpha (+ resid)
// Block tile 256x64, BK=32, 256 threads = 8 waves. Each wave owns TWO 16-row
// strips so every B fragment feeds 2 WMMAs. Double-buffered LDS filled with
// GLOBAL_LOAD_ASYNC_TO_LDS_B128 (ASYNCcnt) overlapping WMMA compute.
// ---------------------------------------------------------------------------
#define BM 256
#define BN 64
#define BK 32
#define ASTR 40
#define WSTR 40

template <int ACT>
__global__ __launch_bounds__(256)
void gemm_bf16(const unsigned short* __restrict__ A,   // [M][K] bf16
               const unsigned short* __restrict__ W,   // [E][K] bf16
               const float* __restrict__ bias,         // [E]
               const float* __restrict__ resid,        // [M][E] f32 or null
               float* __restrict__ outF,               // [M][E] f32 or null
               unsigned short* __restrict__ outB,      // [M][E] bf16 or null
               int M, int K, int E, float alpha) {
  __shared__ unsigned short lA[2][BM * ASTR];  // 2 x 20480 B
  __shared__ unsigned short lW[2][BN * WSTR];  // 2 x  5120 B
  const int tid  = threadIdx.x;
  const int lane = tid & 31;
  const int wave = tid >> 5;
  const int half = lane >> 4;
  const int l15  = lane & 15;
  const int m0 = blockIdx.x * BM;
  const int e0 = blockIdx.y * BN;

  v8f acc[2][4];
  for (int st = 0; st < 2; ++st)
    for (int nt = 0; nt < 4; ++nt)
      for (int j = 0; j < 8; ++j) acc[st][nt][j] = 0.0f;

  // Async-stage one K-tile (A: 1024 16B chunks, W: 256 16B chunks).
  auto stage = [&](int buf, int k0) {
    for (int i = 0; i < 4; ++i) {
      int c = tid * 4 + i;               // 0..1023
      int r = c >> 2, p = c & 3;
      int gr = m0 + r;
      if (gr >= M) gr = M - 1;           // clamp; extra rows feed discarded C rows
      async_b128(A + (size_t)gr * K + k0 + p * 8, &lA[buf][r * ASTR + p * 8]);
    }
    {
      int r = tid >> 2, p = tid & 3;     // E is a multiple of 64
      async_b128(W + (size_t)(e0 + r) * K + k0 + p * 8, &lW[buf][r * WSTR + p * 8]);
    }
  };

  const int nsteps = K / BK;
  stage(0, 0);
  wait_async0();
  __syncthreads();

  for (int i = 0; i < nsteps; ++i) {
    const int buf = i & 1;
    if (i + 1 < nsteps) stage(buf ^ 1, (i + 1) * BK);   // overlap with compute

    Frag16 a0, a1;
    const unsigned short* ap0 = &lA[buf][(wave * 16 + l15) * ASTR];
    const unsigned short* ap1 = ap0 + 128 * ASTR;
    a0.q[0] = *(const v4u*)(ap0 + half * 8);        // K = half*8 .. +7
    a0.q[1] = *(const v4u*)(ap0 + 16 + half * 8);   // K = 16+half*8 .. +7
    a1.q[0] = *(const v4u*)(ap1 + half * 8);
    a1.q[1] = *(const v4u*)(ap1 + 16 + half * 8);

    for (int nt = 0; nt < 4; ++nt) {
      Frag16 b;
      const unsigned short* wp = &lW[buf][(nt * 16 + l15) * WSTR + half * 8];
      b.q[0] = *(const v4u*)(wp);       // K = half*8 .. +7
      b.q[1] = *(const v4u*)(wp + 16);  // K = 16+half*8 .. +7
      acc[0][nt] = wmma_bf16(a0.v, b.v, acc[0][nt]);
      acc[1][nt] = wmma_bf16(a1.v, b.v, acc[1][nt]);
    }
    if (i + 1 < nsteps) wait_async0();  // next buffer landed
    __syncthreads();
  }

  // Epilogue: row = m0 + st*128 + wave*16 + j + 8*half, col = e0 + nt*16 + l15.
  for (int st = 0; st < 2; ++st)
    for (int nt = 0; nt < 4; ++nt) {
      const int col = e0 + nt * 16 + l15;
      const float bv = bias ? bias[col] : 0.0f;
      for (int j = 0; j < 8; ++j) {
        const int row = m0 + st * 128 + wave * 16 + j + 8 * half;
        if (row >= M) continue;
        float v = acc[st][nt][j] + bv;
        if (ACT == 1) v = gelu_exact(v);
        v *= alpha;
        if (resid) v += resid[(size_t)row * E + col];
        if (outF) outF[(size_t)row * E + col] = v;
        if (outB) outB[(size_t)row * E + col] = f2bf(v);
      }
    }
}

// ---------------------------------------------------------------------------
// Attention: one wave per (qtile16, head, batch). qkv bf16 [B*S][2304].
// scores = (Q K^T)/8 via WMMA, register softmax with shfl_xor row reduce,
// P (bf16, LDS) x V (LDS, transposed) via WMMA. Out bf16 [B*S][768].
// ---------------------------------------------------------------------------
__global__ __launch_bounds__(32)
void attn_kernel(const unsigned short* __restrict__ qkv,
                 unsigned short* __restrict__ out) {
  __shared__ unsigned short Kt[SEQ * 72];   // [token][dh], stride 72
  __shared__ unsigned short Vt[64 * 232];   // [dh][token] transposed, stride 232
  __shared__ unsigned short Pt[16 * 224];   // [qrow][token], stride 224
  const int lane = threadIdx.x;
  const int qt = blockIdx.x;   // 0..12
  const int h  = blockIdx.y;
  const int b  = blockIdx.z;
  const int half = lane >> 4, l15 = lane & 15;
  const size_t qbase = (size_t)b * SEQ * DQKV;

  // Zero Vt (covers token padding 197..231 so 0*pad stays finite).
  for (int c = lane; c < 64 * 232 / 8; c += 32) *(v4u*)(Vt + c * 8) = (v4u)0u;
  // Stage K rows via async memory->LDS (4 x 16B chunks per row).
  for (int c = lane; c < SEQ * 4; c += 32) {
    int t = c >> 2, p = c & 3;
    async_b128(qkv + qbase + (size_t)t * DQKV + DMODEL + h * 64 + p * 8,
               &Kt[t * 72 + p * 8]);
  }
  // Stage V transposed: element (t, d) -> Vt[d][t] (scatter stays on ds path).
  for (int c = lane; c < SEQ * 4; c += 32) {
    int t = c >> 2, p = c & 3;
    Chunk8 ch;
    ch.q = *(const v4u*)(qkv + qbase + (size_t)t * DQKV + 2 * DMODEL + h * 64 + p * 8);
    for (int u = 0; u < 8; ++u) Vt[(p * 8 + u) * 232 + t] = ch.s[u];
  }
  // Zero P pad columns 208..223.
  for (int i = lane; i < 16 * 16; i += 32) Pt[(i >> 4) * 224 + 208 + (i & 15)] = 0;
  wait_async0();
  __syncthreads();

  // Q fragments (dh=64 -> two 16x32 A fragments), OOB q rows zeroed.
  Frag16 qa0, qa1;
  {
    const int tq = qt * 16 + l15;
    if (tq < SEQ) {
      const unsigned short* qp = qkv + qbase + (size_t)tq * DQKV + h * 64;
      qa0.q[0] = *(const v4u*)(qp + half * 8);
      qa0.q[1] = *(const v4u*)(qp + 16 + half * 8);
      qa1.q[0] = *(const v4u*)(qp + 32 + half * 8);
      qa1.q[1] = *(const v4u*)(qp + 48 + half * 8);
    } else {
      qa0.q[0] = (v4u)0u; qa0.q[1] = (v4u)0u;
      qa1.q[0] = (v4u)0u; qa1.q[1] = (v4u)0u;
    }
  }

  // Scores: 13 key tiles x (2 WMMA over dh).
  float s[13][8];
  for (int c = 0; c < 13; ++c) {
    v8f a;
    for (int j = 0; j < 8; ++j) a[j] = 0.0f;
    Frag16 kb0, kb1;
    const int tk = c * 16 + l15;
    if (tk < SEQ) {
      const unsigned short* kp = Kt + tk * 72;
      kb0.q[0] = *(const v4u*)(kp + half * 8);
      kb0.q[1] = *(const v4u*)(kp + 16 + half * 8);
      kb1.q[0] = *(const v4u*)(kp + 32 + half * 8);
      kb1.q[1] = *(const v4u*)(kp + 48 + half * 8);
    } else {
      kb0.q[0] = (v4u)0u; kb0.q[1] = (v4u)0u;
      kb1.q[0] = (v4u)0u; kb1.q[1] = (v4u)0u;
    }
    a = wmma_bf16(qa0.v, kb0.v, a);
    a = wmma_bf16(qa1.v, kb1.v, a);
    const bool valid = (tk < SEQ);
    for (int j = 0; j < 8; ++j) s[c][j] = valid ? a[j] * 0.125f : -1.0e30f;
  }

  // Row softmax: row m = j + 8*half, reduce across the 16 lanes of the half.
  float rm[8], rs[8];
  for (int j = 0; j < 8; ++j) {
    float m = s[0][j];
    for (int c = 1; c < 13; ++c) m = fmaxf(m, s[c][j]);
    for (int off = 1; off < 16; off <<= 1) m = fmaxf(m, __shfl_xor(m, off, 32));
    rm[j] = m;
  }
  for (int j = 0; j < 8; ++j) {
    float acc = 0.0f;
    for (int c = 0; c < 13; ++c) {
      float p = __expf(s[c][j] - rm[j]);
      s[c][j] = p;
      acc += p;
    }
    for (int off = 1; off < 16; off <<= 1) acc += __shfl_xor(acc, off, 32);
    rs[j] = 1.0f / acc;
  }
  for (int c = 0; c < 13; ++c)
    for (int j = 0; j < 8; ++j)
      Pt[(j + 8 * half) * 224 + c * 16 + l15] = f2bf(s[c][j] * rs[j]);
  __syncthreads();

  // out = P[16x224] x V[224x64]: 7 K-steps, 4 dh tiles.
  v8f ov[4];
  for (int nt = 0; nt < 4; ++nt)
    for (int j = 0; j < 8; ++j) ov[nt][j] = 0.0f;
  for (int ks = 0; ks < 7; ++ks) {
    Frag16 pa;
    const unsigned short* pp = Pt + l15 * 224 + ks * 32;
    pa.q[0] = *(const v4u*)(pp + half * 8);
    pa.q[1] = *(const v4u*)(pp + 16 + half * 8);
    for (int nt = 0; nt < 4; ++nt) {
      Frag16 vb;
      const unsigned short* vp = Vt + (nt * 16 + l15) * 232 + ks * 32;
      vb.q[0] = *(const v4u*)(vp + half * 8);
      vb.q[1] = *(const v4u*)(vp + 16 + half * 8);
      ov[nt] = wmma_bf16(pa.v, vb.v, ov[nt]);
    }
  }
  const size_t obase = (size_t)b * SEQ * DMODEL;
  for (int nt = 0; nt < 4; ++nt)
    for (int j = 0; j < 8; ++j) {
      const int tq = qt * 16 + j + 8 * half;
      if (tq < SEQ)
        out[obase + (size_t)tq * DMODEL + h * 64 + nt * 16 + l15] = f2bf(ov[nt][j]);
    }
}

// ---------------------------------------------------------------------------
// Host orchestration.
// ---------------------------------------------------------------------------
extern "C" void kernel_launch(void* const* d_in, const int* in_sizes, int n_in,
                              void* d_out, int out_size, void* d_ws, size_t ws_size,
                              hipStream_t stream) {
  (void)in_sizes; (void)n_in; (void)out_size; (void)ws_size;
  const float* x_in   = (const float*)d_in[0];
  const float* w_in   = (const float*)d_in[1];
  const float* b_in   = (const float*)d_in[2];
  const float* w_outp = (const float*)d_in[3];
  const float* b_outp = (const float*)d_in[4];
  const float* lnA_g  = (const float*)d_in[5];
  const float* lnA_b  = (const float*)d_in[6];
  const float* ln1_g  = (const float*)d_in[7];
  const float* ln1_b  = (const float*)d_in[8];
  const float* ln2_g  = (const float*)d_in[9];
  const float* ln2_b  = (const float*)d_in[10];
  const float* wI1 = (const float*)d_in[11];
  const float* bI1 = (const float*)d_in[12];
  const float* wI2 = (const float*)d_in[13];
  const float* bI2 = (const float*)d_in[14];
  const float* wF1 = (const float*)d_in[15];
  const float* bF1 = (const float*)d_in[16];
  const float* wF2 = (const float*)d_in[17];
  const float* bF2 = (const float*)d_in[18];

  char* base = (char*)d_ws;
  size_t off = 0;
  auto alloc = [&](size_t bytes) -> char* {
    off = (off + 255) & ~(size_t)255;
    char* p = base + off;
    off += bytes;
    return p;
  };
  unsigned short* wIn_b  = (unsigned short*)alloc((size_t)DQKV * DMODEL * 2);
  unsigned short* wOut_b = (unsigned short*)alloc((size_t)DMODEL * DMODEL * 2);
  unsigned short* wI1_b  = (unsigned short*)alloc((size_t)DMODEL * DMODEL * 2);
  unsigned short* wI2_b  = (unsigned short*)alloc((size_t)DMODEL * DMODEL * 2);
  unsigned short* wF1_b  = (unsigned short*)alloc((size_t)DFF * DMODEL * 2);
  unsigned short* wF2_b  = (unsigned short*)alloc((size_t)DMODEL * DFF * 2);
  unsigned short* R1   = (unsigned short*)alloc((size_t)NTOK * DMODEL * 2); // xn -> attn -> xn2
  unsigned short* QH   = (unsigned short*)alloc((size_t)NTOK * DFF * 2);    // qkv, later ffn hidden
  unsigned short* st_b = (unsigned short*)alloc((size_t)NTOK * DMODEL * 2);
  unsigned short* h_b  = (unsigned short*)alloc((size_t)NTOK * DMODEL * 2);
  float* x1  = (float*)alloc((size_t)NTOK * DMODEL * 4);
  float* x2  = (float*)alloc((size_t)NTOK * DMODEL * 4);
  float* st0 = (float*)alloc((size_t)NTOK * DMODEL * 4);
  float* st1 = (float*)alloc((size_t)NTOK * DMODEL * 4);

  auto cvt = [&](const float* src, unsigned short* dst, int n) {
    f2bf_kernel<<<(n + 255) / 256, 256, 0, stream>>>(src, dst, n);
  };
  cvt(w_in,   wIn_b,  DQKV * DMODEL);
  cvt(w_outp, wOut_b, DMODEL * DMODEL);
  cvt(wI1,    wI1_b,  DMODEL * DMODEL);
  cvt(wI2,    wI2_b,  DMODEL * DMODEL);
  cvt(wF1,    wF1_b,  DFF * DMODEL);
  cvt(wF2,    wF2_b,  DMODEL * DFF);

  const dim3 gd((NTOK + BM - 1) / BM, DMODEL / BN);
  const dim3 gq((NTOK + BM - 1) / BM, DQKV / BN);
  const dim3 gf((NTOK + BM - 1) / BM, DFF / BN);

  // xn = LN(x); qkv = xn @ Win^T + b
  ln_kernel<<<NTOK, 256, 0, stream>>>(x_in, lnA_g, lnA_b, R1, nullptr);
  gemm_bf16<0><<<gq, 256, 0, stream>>>(R1, wIn_b, b_in, nullptr, nullptr, QH,
                                       NTOK, DMODEL, DQKV, 1.0f);
  // attention -> R1 (bf16)
  attn_kernel<<<dim3(13, NHEAD, BATCH), 32, 0, stream>>>(QH, R1);
  // x1 = x + attn @ Wout^T + b
  gemm_bf16<0><<<gd, 256, 0, stream>>>(R1, wOut_b, b_outp, x_in, x1, nullptr,
                                       NTOK, DMODEL, DMODEL, 1.0f);
  // state0 = LN(x1) (f32 + bf16)
  ln_kernel<<<NTOK, 256, 0, stream>>>(x1, ln1_g, ln1_b, st_b, st0);
  // INL: 5 x { h = GELU(state@W1^T+b1); state = state + 0.1*(h@W2^T+b2) }
  float* sp = st0;
  float* sn = st1;
  for (int it = 0; it < 5; ++it) {
    gemm_bf16<1><<<gd, 256, 0, stream>>>(st_b, wI1_b, bI1, nullptr, nullptr, h_b,
                                         NTOK, DMODEL, DMODEL, 1.0f);
    gemm_bf16<0><<<gd, 256, 0, stream>>>(h_b, wI2_b, bI2, sp, sn, st_b,
                                         NTOK, DMODEL, DMODEL, 0.1f);
    float* t = sp; sp = sn; sn = t;
  }
  // x2 = x1 + state
  add_kernel<<<(NTOK * DMODEL + 255) / 256, 256, 0, stream>>>(x1, sp, x2, NTOK * DMODEL);
  // FFN: out = x2 + GELU(LN(x2)@Wf1^T+b1)@Wf2^T + b2
  ln_kernel<<<NTOK, 256, 0, stream>>>(x2, ln2_g, ln2_b, R1, nullptr);
  gemm_bf16<1><<<gf, 256, 0, stream>>>(R1, wF1_b, bF1, nullptr, nullptr, QH,
                                       NTOK, DMODEL, DFF, 1.0f);
  gemm_bf16<0><<<gd, 256, 0, stream>>>(QH, wF2_b, bF2, x2, (float*)d_out, nullptr,
                                       NTOK, DFF, DMODEL, 1.0f);
}
